// Seq2Seq_63041529971206
// MI455X (gfx1250) — compile-verified, matrix operand order checked by reading, not run on hardware
//
#include <hip/hip_runtime.h>
#include <hip/hip_bf16.h>
#include <hip/hip_fp16.h>

// ---------------- problem dims ----------------
#define B_  32
#define E_  256
#define H_  256
#define G4  1024   // 4*H
#define V_  32000
#define S_  64
#define T_  64

typedef __attribute__((ext_vector_type(16))) _Float16 v16h;
typedef __attribute__((ext_vector_type(8)))  float    v8f;

// ---------------- WMMA helpers ----------------
// 16x16x32 f16 -> f32 accumulate (codegen-confirmed builtin for gfx1250)
__device__ __forceinline__ v8f wmma_f16(v16h a, v16h b, v8f c) {
    return __builtin_amdgcn_wmma_f32_16x16x32_f16(
        /*neg_a=*/false, a, /*neg_b=*/false, b,
        /*c_mod=*/(short)0, c, /*reuse_a=*/false, /*reuse_b=*/false);
}

// Contiguous k order, same for A and B (not the ISA table order): lanes 0-15 hold
// K=k0..k0+15 of row (row0+lane); lanes 16-31 hold K=k0+16..k0+31 of row
// (row0+lane-16). Row-major [rows, ld] storage, K contiguous -> one aligned
// 32-byte contiguous load per lane.
__device__ __forceinline__ v16h ld_tile(const _Float16* base, int row0,
                                        int ld, int k0, int lane) {
    int r = row0 + (lane & 15);
    int k = k0 + (lane & 16);          // +16 for upper half-wave
    return *(const v16h*)(base + (size_t)r * (size_t)ld + k);
}

__device__ __forceinline__ float sigmoidf_(float x) {
    return 1.0f / (1.0f + __expf(-x));
}

// ---------------- shared LSTM gate GEMM ----------------
// gates[32,1024] = x[32,256] @ Wih^T + h[32,256] @ Whh^T
// 512 threads = 16 waves, 8 tiles per wave (2 M-tiles x 64 N-tiles total).
// K-steps are processed in software-pipelined batches of 4 (loads batched
// ahead of the 4-WMMA chain so the scheduler can overlap load latency).
__device__ __forceinline__ void lstm_gate_gemm(const _Float16* x_sh,
                                               const _Float16* h_sh,
                                               const _Float16* Wih,
                                               const _Float16* Whh,
                                               float* gates, int tid) {
    int wid  = tid >> 5;
    int lane = tid & 31;
    for (int tt = 0; tt < 8; ++tt) {
        int tile  = wid * 8 + tt;          // 0..127
        int mtile = tile & 1;              // 0..1   (batch rows /16)
        int ntile = tile >> 1;             // 0..63  (gate cols /16)
        v8f acc = {};
        #pragma unroll
        for (int g = 0; g < 2; ++g) {
            v16h a[4], b[4];
            #pragma unroll
            for (int k = 0; k < 4; ++k) {
                int k0 = (g * 4 + k) * 32;
                a[k] = ld_tile(x_sh, mtile * 16, E_, k0, lane);
                b[k] = ld_tile(Wih,  ntile * 16, E_, k0, lane);
            }
            #pragma unroll
            for (int k = 0; k < 4; ++k) acc = wmma_f16(a[k], b[k], acc);
        }
        #pragma unroll
        for (int g = 0; g < 2; ++g) {
            v16h a[4], b[4];
            #pragma unroll
            for (int k = 0; k < 4; ++k) {
                int k0 = (g * 4 + k) * 32;
                a[k] = ld_tile(h_sh, mtile * 16, H_, k0, lane);
                b[k] = ld_tile(Whh,  ntile * 16, H_, k0, lane);
            }
            #pragma unroll
            for (int k = 0; k < 4; ++k) acc = wmma_f16(a[k], b[k], acc);
        }
        // C tile: reg r, lane l -> M = r + 8*(l>=16), N = l%16
        int ncol  = ntile * 16 + (lane & 15);
        int mbase = mtile * 16 + ((lane >> 4) << 3);
        #pragma unroll
        for (int r = 0; r < 8; ++r)
            gates[(size_t)(mbase + r) * G4 + ncol] = acc[r];
    }
}

// ---------------- embed gather: 32B vector path ----------------
// 512 threads cover 32x256; each thread converts 16 contiguous floats
// into one v16h and does a single 32-byte LDS store.
__device__ __forceinline__ void embed_to_lds(const float* __restrict__ emb,
                                             const int* __restrict__ toks,
                                             int tok_stride,  // elements between batch entries
                                             _Float16* x_sh, int tid) {
    int base = tid * 16;                 // 0..8191
    int b  = base >> 8;
    int e0 = base & 255;
    int tok = toks[b * tok_stride];
    const float4* s4 = (const float4*)(emb + (size_t)tok * E_ + e0);
    _Float16 tmp[16];
    #pragma unroll
    for (int j = 0; j < 4; ++j) {
        float4 f = s4[j];
        tmp[j * 4 + 0] = (_Float16)f.x;
        tmp[j * 4 + 1] = (_Float16)f.y;
        tmp[j * 4 + 2] = (_Float16)f.z;
        tmp[j * 4 + 3] = (_Float16)f.w;
    }
    *(v16h*)(x_sh + base) = *(const v16h*)tmp;
}

// ---------------- prep: f32 -> f16 (vectorized) ----------------
__global__ void __launch_bounds__(256) cvt_f16_kernel(const float* __restrict__ s,
                                                      _Float16* __restrict__ d,
                                                      int n4) {   // n/4
    int i = blockIdx.x * 256 + threadIdx.x;
    if (i < n4) {
        float4 f = ((const float4*)s)[i];
        _Float16 h4[4] = {(_Float16)f.x, (_Float16)f.y, (_Float16)f.z, (_Float16)f.w};
        ((uint2*)d)[i] = *(const uint2*)h4;
    }
}

__global__ void __launch_bounds__(256) zero_f32_kernel(float4* __restrict__ p, int n4) {
    int i = blockIdx.x * 256 + threadIdx.x;
    if (i < n4) p[i] = make_float4(0.f, 0.f, 0.f, 0.f);
}

// ---------------- encoder: 64 LSTM steps in one persistent block ----------------
__global__ void __launch_bounds__(512)
enc_kernel(const int* __restrict__ src, const int* __restrict__ trg,
           const float* __restrict__ emb,
           const _Float16* __restrict__ Wih, const _Float16* __restrict__ Whh,
           const float* __restrict__ bih, const float* __restrict__ bhh,
           _Float16* __restrict__ h16_out, float* __restrict__ c_out,
           int* __restrict__ tok_out) {
    extern __shared__ char smem[];
    _Float16* x_sh = (_Float16*)smem;                 // 32*256 f16 = 16 KB
    _Float16* h_sh = (_Float16*)(smem + 16384);       // 32*256 f16 = 16 KB
    float*    gates = (float*)(smem + 32768);         // 32*1024 f32 = 128 KB
    int tid = threadIdx.x;

    ((v16h*)h_sh)[tid] = (v16h)(_Float16)0.0f;        // 512*16 = 8192 halves
    float c_reg[16];
    #pragma unroll
    for (int q = 0; q < 16; ++q) c_reg[q] = 0.0f;
    __syncthreads();

    for (int t = 0; t < S_; ++t) {
        embed_to_lds(emb, src + t * B_, 1, x_sh, tid);
        __syncthreads();
        lstm_gate_gemm(x_sh, h_sh, Wih, Whh, gates, tid);
        __syncthreads();
        // activations: thread owns 16 fixed (b,u) cells; c stays in registers
        _Float16 htmp[16];
        #pragma unroll
        for (int q = 0; q < 16; ++q) {
            int idx = tid * 16 + q;           // 0..8191
            int b = idx >> 8, u = idx & 255;
            const float* g = gates + (size_t)b * G4;
            float iv = g[u]       + bih[u]       + bhh[u];
            float fv = g[256 + u] + bih[256 + u] + bhh[256 + u];
            float gv = g[512 + u] + bih[512 + u] + bhh[512 + u];
            float ov = g[768 + u] + bih[768 + u] + bhh[768 + u];
            float cn = sigmoidf_(fv) * c_reg[q] + sigmoidf_(iv) * tanhf(gv);
            float hn = sigmoidf_(ov) * tanhf(cn);
            c_reg[q] = cn;
            htmp[q] = (_Float16)hn;
        }
        ((v16h*)h_sh)[tid] = *(const v16h*)htmp;
        // next iteration's __syncthreads (after x write) orders h_sh/gates reuse
    }
    __syncthreads();
    ((v16h*)h16_out)[tid] = ((v16h*)h_sh)[tid];
    #pragma unroll
    for (int q = 0; q < 16; ++q) c_out[tid * 16 + q] = c_reg[q];
    if (tid < B_) tok_out[tid] = trg[tid];    // trg[0][b]
}

// ---------------- decoder LSTM: one step ----------------
__global__ void __launch_bounds__(512)
dec_lstm_kernel(const int* __restrict__ tok_in, const float* __restrict__ emb,
                const _Float16* __restrict__ Wih, const _Float16* __restrict__ Whh,
                const float* __restrict__ bih, const float* __restrict__ bhh,
                _Float16* __restrict__ h16, float* __restrict__ c) {
    extern __shared__ char smem[];
    _Float16* x_sh = (_Float16*)smem;
    _Float16* h_sh = (_Float16*)(smem + 16384);
    float*    gates = (float*)(smem + 32768);
    int tid = threadIdx.x;

    embed_to_lds(emb, tok_in, 1, x_sh, tid);
    ((v16h*)h_sh)[tid] = ((const v16h*)h16)[tid];
    __syncthreads();
    lstm_gate_gemm(x_sh, h_sh, Wih, Whh, gates, tid);
    __syncthreads();
    _Float16 htmp[16];
    #pragma unroll
    for (int q = 0; q < 16; ++q) {
        int idx = tid * 16 + q;
        int b = idx >> 8, u = idx & 255;
        const float* g = gates + (size_t)b * G4;
        float iv = g[u]       + bih[u]       + bhh[u];
        float fv = g[256 + u] + bih[256 + u] + bhh[256 + u];
        float gv = g[512 + u] + bih[512 + u] + bhh[512 + u];
        float ov = g[768 + u] + bih[768 + u] + bhh[768 + u];
        float cn = sigmoidf_(fv) * c[idx] + sigmoidf_(iv) * tanhf(gv);
        float hn = sigmoidf_(ov) * tanhf(cn);
        c[idx] = cn;
        htmp[q] = (_Float16)hn;
    }
    ((v16h*)h16)[tid] = *(const v16h*)htmp;
}

// ---------------- FC: logits[32,32000] = h @ fc_W^T + b ----------------
// 250 blocks x 256 threads (8 waves). Each wave computes one 16-col N-tile
// for BOTH 16-row M-tiles: every B fragment (fc_W, the dominant L2 stream)
// loaded once feeds two WMMAs. K pipelined in batches of 4.
__global__ void __launch_bounds__(256)
fc_kernel(const _Float16* __restrict__ h16, const _Float16* __restrict__ W,
          const float* __restrict__ bias, float* __restrict__ out) {
    int tid   = threadIdx.x;
    int wid   = tid >> 5;
    int lane  = tid & 31;
    int nbase = blockIdx.x * 128 + wid * 16;   // this wave's 16 vocab columns
    v8f acc0 = {}, acc1 = {};
    #pragma unroll
    for (int g = 0; g < 2; ++g) {
        v16h a0[4], a1[4], b[4];
        #pragma unroll
        for (int k = 0; k < 4; ++k) {
            int k0 = (g * 4 + k) * 32;
            b[k]  = ld_tile(W, nbase, H_, k0, lane);
            a0[k] = ld_tile(h16,  0, H_, k0, lane);
            a1[k] = ld_tile(h16, 16, H_, k0, lane);
        }
        if (g == 0)
            __builtin_prefetch(W + (size_t)(nbase + (lane & 15)) * H_ + 128, 0, 0);
        #pragma unroll
        for (int k = 0; k < 4; ++k) {
            acc0 = wmma_f16(a0[k], b[k], acc0);
            acc1 = wmma_f16(a1[k], b[k], acc1);
        }
    }
    int ncol  = nbase + (lane & 15);
    int mbase = (lane >> 4) << 3;              // 0 or 8
    float bb = bias[ncol];
    #pragma unroll
    for (int r = 0; r < 8; ++r) {
        out[(size_t)(mbase + r) * V_ + ncol]      = acc0[r] + bb;
        out[(size_t)(mbase + r + 16) * V_ + ncol] = acc1[r] + bb;
    }
}

// ---------------- per-row stats: max/argmax, log-sum-exp, token select ----------------
__global__ void __launch_bounds__(256)
rowstats_kernel(const float* __restrict__ logits, const int* __restrict__ trg_row,
                const int* __restrict__ tf, float* __restrict__ rowmax,
                float* __restrict__ rowlse, int* __restrict__ tok) {
    __shared__ float smax[256];
    __shared__ int   sarg[256];
    __shared__ float ssum[256];
    int b = blockIdx.x, tid = threadIdx.x;
    const float4* row4 = (const float4*)(logits + (size_t)b * V_);
    const int nV4 = V_ / 4;                    // 8000
    float m = -__builtin_inff(); int arg = 0;
    for (int v4 = tid; v4 < nV4; v4 += 256) {
        float4 f = row4[v4];
        float vals[4] = {f.x, f.y, f.z, f.w};
        #pragma unroll
        for (int j = 0; j < 4; ++j)
            if (vals[j] > m) { m = vals[j]; arg = v4 * 4 + j; }
    }
    smax[tid] = m; sarg[tid] = arg;
    __syncthreads();
    for (int s2 = 128; s2 > 0; s2 >>= 1) {
        if (tid < s2) {
            float om = smax[tid + s2]; int oa = sarg[tid + s2];
            if (om > smax[tid] || (om == smax[tid] && oa < sarg[tid])) {
                smax[tid] = om; sarg[tid] = oa;
            }
        }
        __syncthreads();
    }
    float M = smax[0];
    float s = 0.0f;
    for (int v4 = tid; v4 < nV4; v4 += 256) {
        float4 f = row4[v4];
        s += __expf(f.x - M) + __expf(f.y - M) + __expf(f.z - M) + __expf(f.w - M);
    }
    ssum[tid] = s;
    __syncthreads();
    for (int s2 = 128; s2 > 0; s2 >>= 1) {
        if (tid < s2) ssum[tid] += ssum[tid + s2];
        __syncthreads();
    }
    if (tid == 0) {
        rowmax[b] = M;
        rowlse[b] = __logf(ssum[0]);
        tok[b] = (tf[0] != 0) ? trg_row[b] : sarg[0];
    }
}

// ---------------- in-place log-softmax normalize (vectorized) ----------------
__global__ void __launch_bounds__(256)
norm_kernel(float4* __restrict__ logits, const float* __restrict__ rowmax,
            const float* __restrict__ rowlse) {
    int i = blockIdx.x * 256 + threadIdx.x;    // B_*V_/4 = 256000
    if (i < B_ * V_ / 4) {
        int b = i / (V_ / 4);
        float d = rowmax[b] + rowlse[b];
        float4 f = logits[i];
        f.x -= d; f.y -= d; f.z -= d; f.w -= d;
        logits[i] = f;
    }
}

// ---------------- host launch ----------------
extern "C" void kernel_launch(void* const* d_in, const int* in_sizes, int n_in,
                              void* d_out, int out_size, void* d_ws, size_t ws_size,
                              hipStream_t stream) {
    const int*   src      = (const int*)d_in[0];
    const int*   trg      = (const int*)d_in[1];
    const int*   tf_mask  = (const int*)d_in[2];
    const float* enc_emb  = (const float*)d_in[3];
    const float* enc_W_ih = (const float*)d_in[4];
    const float* enc_W_hh = (const float*)d_in[5];
    const float* enc_b_ih = (const float*)d_in[6];
    const float* enc_b_hh = (const float*)d_in[7];
    const float* dec_emb  = (const float*)d_in[8];
    const float* dec_W_ih = (const float*)d_in[9];
    const float* dec_W_hh = (const float*)d_in[10];
    const float* dec_b_ih = (const float*)d_in[11];
    const float* dec_b_hh = (const float*)d_in[12];
    const float* fc_W     = (const float*)d_in[13];
    const float* fc_b     = (const float*)d_in[14];
    float* out = (float*)d_out;

    // workspace layout
    char* ws = (char*)d_ws;
    size_t off = 0;
    _Float16* eWih  = (_Float16*)(ws + off); off += (size_t)G4 * E_ * 2;
    _Float16* eWhh  = (_Float16*)(ws + off); off += (size_t)G4 * H_ * 2;
    _Float16* dWih  = (_Float16*)(ws + off); off += (size_t)G4 * E_ * 2;
    _Float16* dWhh  = (_Float16*)(ws + off); off += (size_t)G4 * H_ * 2;
    _Float16* fcW16 = (_Float16*)(ws + off); off += (size_t)V_ * H_ * 2;
    _Float16* h16   = (_Float16*)(ws + off); off += (size_t)B_ * H_ * 2;
    float*    c32   = (float*)(ws + off);    off += (size_t)B_ * H_ * 4;
    int*      tok   = (int*)(ws + off);      off += 128;
    float*    rmax  = (float*)(ws + off);    off += 128;
    float*    rlse  = (float*)(ws + off);    off += 128;

    const int nW4 = G4 * E_ / 4;   // 65536
    const int nF4 = V_ * H_ / 4;   // 2,048,000
    cvt_f16_kernel<<<(nW4 + 255) / 256, 256, 0, stream>>>(enc_W_ih, eWih, nW4);
    cvt_f16_kernel<<<(nW4 + 255) / 256, 256, 0, stream>>>(enc_W_hh, eWhh, nW4);
    cvt_f16_kernel<<<(nW4 + 255) / 256, 256, 0, stream>>>(dec_W_ih, dWih, nW4);
    cvt_f16_kernel<<<(nW4 + 255) / 256, 256, 0, stream>>>(dec_W_hh, dWhh, nW4);
    cvt_f16_kernel<<<(nF4 + 255) / 256, 256, 0, stream>>>(fc_W, fcW16, nF4);

    // outputs[0] = zeros
    const int nO4 = B_ * V_ / 4;   // 256000
    zero_f32_kernel<<<(nO4 + 255) / 256, 256, 0, stream>>>((float4*)out, nO4);

    const size_t lds_bytes = 16384 + 16384 + (size_t)B_ * G4 * 4;  // 160 KB

    // encoder: 64 steps, one persistent block
    enc_kernel<<<1, 512, lds_bytes, stream>>>(src, trg, enc_emb, eWih, eWhh,
                                              enc_b_ih, enc_b_hh, h16, c32, tok);

    // decoder: 63 autoregressive steps
    for (int s = 0; s < T_ - 1; ++s) {
        float* out_step = out + (size_t)(s + 1) * B_ * V_;
        dec_lstm_kernel<<<1, 512, lds_bytes, stream>>>(tok, dec_emb, dWih, dWhh,
                                                       dec_b_ih, dec_b_hh, h16, c32);
        fc_kernel<<<V_ / 128, 256, 0, stream>>>(h16, fcW16, fc_b, out_step);
        rowstats_kernel<<<B_, 256, 0, stream>>>(out_step, trg + (size_t)(s + 1) * B_,
                                                tf_mask + (s + 1), rmax, rlse, tok);
        norm_kernel<<<(nO4 + 255) / 256, 256, 0, stream>>>((float4*)out_step, rmax, rlse);
    }
}
